// LLPredRigidityMACE_60559038874214
// MI455X (gfx1250) — matrix operand order repaired and checked by reading, not compiled, not run on hardware
//
#include <hip/hip_runtime.h>
#include <hip/hip_bf16.h>

// ---------------------------------------------------------------------------
// MACE-like GNN energy/forces for MI455X (gfx1250, wave32, WMMA 16x16x32 f16)
// TDM (tensor_load_to_lds) stages edge-feature tiles into LDS for the fused
// WMMA kernels; silu uses hardware v_rcp_f32.
// ---------------------------------------------------------------------------

typedef _Float16 h16;
typedef __attribute__((ext_vector_type(16))) _Float16 v16h;
typedef __attribute__((ext_vector_type(8)))  float    v8f;
typedef __attribute__((ext_vector_type(4)))  float    v4f;
typedef unsigned int v4u __attribute__((ext_vector_type(4)));
typedef int          v8i __attribute__((ext_vector_type(8)));
typedef int          v4i __attribute__((ext_vector_type(4)));

#define NEDGE   256000
#define NNODE   10000
#define NGRAPH  32
#define RMAXC   5.0f
#define INV_AVG (1.0f/25.6f)
#define PI_F    3.14159265358979323846f

#define GF_OUTF16 1
#define GF_ADD    2
#define GF_SILUA  4
#define GF_GRADZ  8

__device__ __forceinline__ float siluf(float x){
  return x * __builtin_amdgcn_rcpf(1.f + __expf(-x));
}
__device__ __forceinline__ float silugf(float x){
  float s = __builtin_amdgcn_rcpf(1.f + __expf(-x));
  return s*(1.f + x*(1.f-s));
}

union Frag { v16h v; v4f q[2]; };

// A fragment: 16x32 f16 tile, row-major A with leading dim lda (halfs).
__device__ __forceinline__ v16h load_a16(const h16* A, int lda, int row, int hh, int kb){
  const h16* p = A + (size_t)row*lda + kb + hh*8;
  Frag f; f.q[0] = *(const v4f*)p; f.q[1] = *(const v4f*)(p + 16); return f.v;
}
// B fragment from Bt (B transposed, [N x K] row-major).
__device__ __forceinline__ v16h load_b16(const h16* Bt, int ldb, int col, int hh, int kb){
  const h16* p = Bt + (size_t)col*ldb + kb + hh*8;
  Frag f; f.q[0] = *(const v4f*)p; f.q[1] = *(const v4f*)(p + 16); return f.v;
}
__device__ __forceinline__ v16h silu16(v16h a){
  v16h r;
  #pragma unroll
  for (int i=0;i<16;i++) r[i] = (h16)siluf((float)a[i]);
  return r;
}
__device__ __forceinline__ v8f wmma_f16(v16h a, v16h b, v8f c){
  return __builtin_amdgcn_wmma_f32_16x16x32_f16(false, a, false, b, (short)0, c, false, false);
}

// ---------------------------------------------------------------------------
// TDM: DMA a 2-D f16 tile (tile_d1 rows x tile_d0 halfs) from global into LDS.
// Issued by one wave; waits TENSORcnt==0 before returning.
// D# per CDNA5 ISA ch.8: group0 = {count,lds_addr,global_addr,type=2},
// group1 = {wgmask=0,data_size=2B, tensor dims/strides, tile dims}.
// ---------------------------------------------------------------------------
__device__ __forceinline__ void tdm_load_2d(const h16* g, h16* lds_dst,
                                            unsigned tile_d0, unsigned tile_d1,
                                            unsigned tensor_d0, unsigned tensor_d1,
                                            unsigned stride0)
{
#if __has_builtin(__builtin_amdgcn_tensor_load_to_lds) && __has_builtin(__builtin_amdgcn_s_wait_tensorcnt)
  unsigned ldsa = (unsigned)(size_t)lds_dst;           // low 32 bits = LDS offset
  unsigned long long ga = (unsigned long long)(size_t)g;
  v4u g0;
  g0[0] = 1u;                                          // count=1 (valid user D#)
  g0[1] = ldsa;                                        // lds_addr
  g0[2] = (unsigned)(ga & 0xFFFFFFFFu);                // global_addr[31:0]
  g0[3] = (unsigned)((ga >> 32) & 0x01FFFFFFu) | (2u << 30);  // addr[56:32] | type=2
  v8i g1;
  g1[0] = (int)(1u << 16);                             // wgmask=0, data_size=1 (2 bytes)
  g1[1] = (int)((tensor_d0 & 0xFFFFu) << 16);          // tensor_dim0[15:0] @bit48
  g1[2] = (int)(((tensor_d0 >> 16) & 0xFFFFu) | ((tensor_d1 & 0xFFFFu) << 16));
  g1[3] = (int)(((tensor_d1 >> 16) & 0xFFFFu) | ((tile_d0 & 0xFFFFu) << 16)); // tile_dim0
  g1[4] = (int)(tile_d1 & 0xFFFFu);                    // tile_dim1 | tile_dim2=0
  g1[5] = (int)stride0;                                // tensor_dim0_stride[31:0]
  g1[6] = 0;                                           // stride hi / dim1_stride
  g1[7] = 0;
  v4i z4 = {0,0,0,0};
#if defined(__clang_major__) && (__clang_major__ >= 23)
  v8i z8 = {0,0,0,0,0,0,0,0};
  __builtin_amdgcn_tensor_load_to_lds(g0, g1, z4, z4, z8, 0);
#else
  __builtin_amdgcn_tensor_load_to_lds(g0, g1, z4, z4, 0);
#endif
  __builtin_amdgcn_s_wait_tensorcnt(0);
#else
  int lane = threadIdx.x & 31;
  int n = (int)(tile_d0 * tile_d1);
  for (int i = lane; i < n; i += 32){
    int rr = i / (int)tile_d0, cc = i - (int)tile_d0 * rr;
    lds_dst[rr*(int)tile_d0 + cc] = g[(size_t)rr*stride0 + cc];
  }
#endif
}

// ---------------------------------------------------------------------------
// Generic WMMA GEMM: D[M x N] = A[M x K] (f16, opt silu-on-load) @ Bt^T
// ---------------------------------------------------------------------------
__global__ void k_gemm(const h16* __restrict__ A, const h16* __restrict__ Bt,
                       void* __restrict__ Out, const h16* __restrict__ Zg,
                       int M, int K, int N, int lda, int ldb,
                       int ldo, int oes, int ooff, int flags)
{
  int lane = threadIdx.x & 31;
  int wave = threadIdx.x >> 5;
  int tile = blockIdx.x * (blockDim.x >> 5) + wave;
  int ntn  = N >> 4;
  int total = (M >> 4) * ntn;
  if (tile >= total) return;
  int mt = tile / ntn, nt = tile - mt*ntn;
  int m0 = mt << 4, n0 = nt << 4;
  int nl = lane & 15, hh = lane >> 4;

  v8f acc = {0.f,0.f,0.f,0.f,0.f,0.f,0.f,0.f};
  for (int kb = 0; kb < K; kb += 32){
    __builtin_prefetch((const void*)(A + (size_t)(m0+nl)*lda + kb + 32), 0, 0);
    v16h a = load_a16(A, lda, m0 + nl, hh, kb);
    if (flags & GF_SILUA) a = silu16(a);
    v16h b = load_b16(Bt, ldb, n0 + nl, hh, kb);
    acc = wmma_f16(a, b, acc);
  }
  #pragma unroll
  for (int v=0; v<8; v++){
    int m = m0 + v + 8*hh;
    int n = n0 + nl;
    float d = acc[v];
    if (flags & GF_GRADZ) d *= silugf((float)Zg[(size_t)m*N + n]);
    if (flags & GF_OUTF16){
      ((h16*)Out)[(size_t)m*ldo + n] = (h16)d;
    } else {
      float* o = (float*)Out + (size_t)m*ldo + (size_t)n*oes + ooff;
      if (flags & GF_ADD) *o += d; else *o = d;
    }
  }
}

// ---------------------------------------------------------------------------
// Forward fused: TDM-stage z3 tile -> tw = silu(z3)@W4 (WMMA) -> msg -> scatter
// ---------------------------------------------------------------------------
__global__ void k_msgagg(const h16* __restrict__ z3, const h16* __restrict__ Bt4,
                         const h16* __restrict__ nf, const float* __restrict__ sh,
                         const int* __restrict__ ei, float* __restrict__ agg, int E)
{
  __shared__ __align__(16) h16 sz3[16*64];
  int lane = threadIdx.x & 31, wave = threadIdx.x >> 5;
  int nl = lane & 15, hh = lane >> 4;
  for (int t = blockIdx.x; t < (E >> 4); t += gridDim.x){
    int e0 = t << 4;
    if (threadIdx.x < 32)
      tdm_load_2d(z3 + (size_t)e0*64, sz3, 64, 16, 64, (unsigned)E, 64);
    __syncthreads();
    for (int nt = wave; nt < 36; nt += 4){
      v8f acc = {0.f,0.f,0.f,0.f,0.f,0.f,0.f,0.f};
      for (int kb = 0; kb < 64; kb += 32){
        v16h a = silu16(load_a16(sz3, 64, nl, hh, kb));
        v16h b = load_b16(Bt4, 64, nt*16 + nl, hh, kb);
        acc = wmma_f16(a, b, acc);
      }
      #pragma unroll
      for (int v=0; v<8; v++){
        int e = e0 + v + 8*hh;
        int c = nt*16 + nl;
        int f = c/9, l = c - 9*f;
        int s = ei[e], rc = ei[E + e];
        float val = acc[v] * (float)nf[(size_t)s*64 + f] * sh[(size_t)e*9 + l] * INV_AVG;
        atomicAdd(&agg[(size_t)rc*576 + f*9 + l], val);
      }
    }
    __syncthreads();
  }
}

// ---------------------------------------------------------------------------
// Backward fused edge kernel:
//  TDM-stage z3 tile; recompute tw; d_tw -> LDS; dsh/dsN atomics;
//  dz3 = (d_tw @ W4^T) * silu'(z3)   (A fed from LDS)
// ---------------------------------------------------------------------------
__global__ void k_edgebwd(const h16* __restrict__ z3, const h16* __restrict__ Bt4,
                          const h16* __restrict__ W4rm, const h16* __restrict__ nf,
                          const float* __restrict__ sh, const float* __restrict__ dagg,
                          const int* __restrict__ ei, float* __restrict__ dsh,
                          float* dsN, h16* __restrict__ dz3, int E)
{
  __shared__ __align__(16) h16 sdtw[16*576];
  __shared__ __align__(16) h16 sz3[16*64];
  int lane = threadIdx.x & 31, wave = threadIdx.x >> 5;
  int nl = lane & 15, hh = lane >> 4;
  for (int t = blockIdx.x; t < (E >> 4); t += gridDim.x){
    int e0 = t << 4;
    if (threadIdx.x < 32)
      tdm_load_2d(z3 + (size_t)e0*64, sz3, 64, 16, 64, (unsigned)E, 64);
    __syncthreads();
    for (int nt = wave; nt < 36; nt += 4){
      v8f acc = {0.f,0.f,0.f,0.f,0.f,0.f,0.f,0.f};
      for (int kb = 0; kb < 64; kb += 32){
        v16h a = silu16(load_a16(sz3, 64, nl, hh, kb));
        v16h b = load_b16(Bt4, 64, nt*16 + nl, hh, kb);
        acc = wmma_f16(a, b, acc);
      }
      #pragma unroll
      for (int v=0; v<8; v++){
        int m = v + 8*hh;
        int e = e0 + m;
        int c = nt*16 + nl;
        int f = c/9, l = c - 9*f;
        int s = ei[e], rc = ei[E + e];
        float g   = dagg[(size_t)rc*576 + f*9 + l] * INV_AVG;
        float nfv = (float)nf[(size_t)s*64 + f];
        float shv = sh[(size_t)e*9 + l];
        float tw  = acc[v];
        sdtw[m*576 + c] = (h16)(g * nfv * shv);
        atomicAdd(&dsh[(size_t)e*9 + l], g * nfv * tw);
        if (dsN) atomicAdd(&dsN[(size_t)s*64 + f], g * shv * tw);
      }
    }
    __syncthreads();
    { // dh3 = sdtw[16x576] @ W4^T  (Bt = W4 row-major 64x576)
      int nt = wave;                    // 4 waves cover N=64
      v8f acc = {0.f,0.f,0.f,0.f,0.f,0.f,0.f,0.f};
      for (int kb = 0; kb < 576; kb += 32){
        v16h a = load_a16(sdtw, 576, nl, hh, kb);
        v16h b = load_b16(W4rm, 576, nt*16 + nl, hh, kb);
        acc = wmma_f16(a, b, acc);
      }
      #pragma unroll
      for (int v=0; v<8; v++){
        int m = v + 8*hh;
        int e = e0 + m;
        int col = nt*16 + nl;
        float d = acc[v] * silugf((float)sz3[m*64 + col]);
        dz3[(size_t)e*64 + col] = (h16)d;
      }
    }
    __syncthreads();
  }
}

// ------------------------------- elementwise -------------------------------

__global__ void k_zero(float* p, long long n){
  long long i = (long long)blockIdx.x*blockDim.x + threadIdx.x;
  if (i < n) p[i] = 0.f;
}
__global__ void k_cvt(const float* s, h16* d, int n){
  int i = blockIdx.x*blockDim.x + threadIdx.x;
  if (i < n) d[i] = (h16)s[i];
}
__global__ void k_cvt_t(const float* s, h16* d, int R, int C, int dR){
  int i = blockIdx.x*blockDim.x + threadIdx.x;
  if (i >= C*dR) return;
  int c = i / dR, r = i - c*dR;
  d[i] = (r < R) ? (h16)s[(size_t)r*C + c] : (h16)0.f;
}
__global__ void k_cvt_padR(const float* s, h16* d, int R, int C, int dR){
  int i = blockIdx.x*blockDim.x + threadIdx.x;
  if (i >= dR*C) return;
  int r = i / C, c = i - r*C;
  d[i] = (r < R) ? (h16)s[(size_t)r*C + c] : (h16)0.f;
}

__global__ void k_geom(const float* __restrict__ pos, const float* __restrict__ shifts,
                       const int* __restrict__ ei, float* __restrict__ sh,
                       float* __restrict__ ru, h16* __restrict__ efp, int E)
{
  int e = blockIdx.x*blockDim.x + threadIdx.x;
  if (e >= E) return;
  int s = ei[e], rc = ei[E + e];
  float vx = pos[rc*3+0]-pos[s*3+0]+shifts[e*3+0];
  float vy = pos[rc*3+1]-pos[s*3+1]+shifts[e*3+1];
  float vz = pos[rc*3+2]-pos[s*3+2]+shifts[e*3+2];
  float r  = sqrtf(vx*vx+vy*vy+vz*vz + 1e-12f);
  float iv = 1.f/r;
  float x = vx*iv, y = vy*iv, z = vz*iv;
  const float s3 = 1.7320508075688772f, s5 = 2.2360679774997896f, s15 = 3.872983346207417f;
  float* S = sh + (size_t)e*9;
  S[0]=1.f; S[1]=s3*x; S[2]=s3*y; S[3]=s3*z;
  S[4]=s15*x*y; S[5]=s15*y*z; S[6]=0.5f*s5*(3.f*z*z-1.f);
  S[7]=s15*x*z; S[8]=0.5f*s15*(x*x-y*y);
  ru[(size_t)e*4+0]=r; ru[(size_t)e*4+1]=x; ru[(size_t)e*4+2]=y; ru[(size_t)e*4+3]=z;
  float xr = r / RMAXC, env = 0.f;
  if (xr < 1.f){
    float x2=xr*xr, x3=x2*xr, x6=x3*x3, x7=x6*xr, x8=x7*xr;
    env = 1.f - 28.f*x6 + 48.f*x7 - 21.f*x8;
  }
  float cc = sqrtf(2.f/RMAXC);
  h16* ef = efp + (size_t)e*32;
  #pragma unroll
  for (int b=0;b<8;b++){
    float nn = (float)(b+1);
    ef[b] = (h16)(cc*__sinf(nn*PI_F*r/RMAXC)*iv*env);
  }
  #pragma unroll
  for (int b=8;b<32;b++) ef[b] = (h16)0.f;
}

__global__ void k_embed(const float* __restrict__ attrs, const float* __restrict__ Wemb,
                        const float* __restrict__ ae, const int* __restrict__ batch,
                        h16* __restrict__ h0, float* __restrict__ e0n,
                        float* __restrict__ totE, int NN)
{
  int i = blockIdx.x*blockDim.x + threadIdx.x;
  if (i >= NN*64) return;
  int n = i >> 6, f = i & 63;
  float s = 0.f;
  for (int k=0;k<10;k++) s += attrs[n*10+k]*Wemb[k*64+f];
  h0[i] = (h16)s;
  if (f == 0){
    float e0 = 0.f;
    for (int k=0;k<10;k++) e0 += attrs[n*10+k]*ae[k];
    e0n[n] = e0;
    atomicAdd(&totE[batch[n]], e0);
  }
}

__global__ void k_aggT(const float* agg, h16* aggT, int NN){
  int i = blockIdx.x*blockDim.x + threadIdx.x;
  if (i >= NN*576) return;
  int n = i / 576, c = i - n*576;
  int f = c/9, l = c - 9*f;
  aggT[(size_t)l*NN*64 + n*64 + f] = (h16)agg[i];
}

__global__ void k_inv(const float* feats, h16* invh, int NN){
  int i = blockIdx.x*blockDim.x + threadIdx.x;
  if (i >= NN*64) return;
  int n = i >> 6, f = i & 63;
  const float* p = feats + (size_t)n*576 + f*9;
  float s = 0.f;
  #pragma unroll
  for (int l=0;l<9;l++) s += p[l]*p[l];
  invh[i] = (h16)s;
}

__global__ void k_hprod(const float* feats, const float* p, float* nfo,
                        h16* sOut, int NN, int layerOff){
  int i = blockIdx.x*blockDim.x + threadIdx.x;
  if (i >= NN*64) return;
  int n = i >> 6, f = i & 63;
  float sc = 1.f + p[i];
  const float* fp = feats + (size_t)n*576 + f*9;
  float* o = nfo + (size_t)n*1152 + layerOff + f*9;
  #pragma unroll
  for (int l=0;l<9;l++) o[l] = fp[l]*sc;
  sOut[i] = (h16)(fp[0]*sc);
}

__global__ void k_read1(const h16* s1, const float* wr, const int* batch,
                        float* e1n, float* totE, int NN){
  int n = blockIdx.x*blockDim.x + threadIdx.x;
  if (n >= NN) return;
  float s = 0.f;
  for (int f=0;f<64;f++) s += (float)s1[(size_t)n*64+f]*wr[f];
  e1n[n] = s;
  atomicAdd(&totE[batch[n]], s);
}

__global__ void k_read2(const float* u, const float* wr2, const float* e0n,
                        const float* e1n, const int* batch, float* e2n,
                        float* nodeE, float* totE, int NN){
  int n = blockIdx.x*blockDim.x + threadIdx.x;
  if (n >= NN) return;
  float s = 0.f;
  #pragma unroll
  for (int k=0;k<16;k++) s += siluf(u[(size_t)n*16+k])*wr2[k];
  e2n[n] = s;
  nodeE[n] = e0n[n] + e1n[n] + s;
  atomicAdd(&totE[batch[n]], s);
}

__global__ void k_bwd_read(const float* u, const float* wr2, const float* Wr1,
                           const float* feats2, float* g, h16* dp, int NN){
  int i = blockIdx.x*blockDim.x + threadIdx.x;
  if (i >= NN*64) return;
  int n = i >> 6, f = i & 63;
  float s = 0.f;
  #pragma unroll
  for (int k=0;k<16;k++) s += wr2[k]*silugf(u[(size_t)n*16+k])*Wr1[f*16+k];
  g[i] = s;
  dp[i] = (h16)(s * feats2[(size_t)n*576 + f*9]);
}

__global__ void k_g1(const float* ds1, const float* wr, const float* feats1,
                     float* g, h16* dp, int NN){
  int i = blockIdx.x*blockDim.x + threadIdx.x;
  if (i >= NN*64) return;
  int n = i >> 6, f = i & 63;
  float s = ds1[i] + wr[f];
  g[i] = s;
  dp[i] = (h16)(s * feats1[(size_t)n*576 + f*9]);
}

__global__ void k_dfeats(const float* feats, const float* p, const float* g,
                         const float* dinv, h16* dfT, int NN){
  int i = blockIdx.x*blockDim.x + threadIdx.x;
  if (i >= NN*64) return;
  int n = i >> 6, f = i & 63;
  float di = dinv[i], gf = g[i], sc = 1.f + p[i];
  const float* fp = feats + (size_t)n*576 + f*9;
  #pragma unroll
  for (int l=0;l<9;l++){
    float d = 2.f*fp[l]*di + ((l==0) ? gf*sc : 0.f);
    dfT[(size_t)l*NN*64 + n*64 + f] = (h16)d;
  }
}

__global__ void k_force(const float* __restrict__ ru, const float* __restrict__ dsh,
                        const float* __restrict__ defv, const int* __restrict__ ei,
                        float* __restrict__ F, int E){
  int e = blockIdx.x*blockDim.x + threadIdx.x;
  if (e >= E) return;
  float r = ru[(size_t)e*4], x = ru[(size_t)e*4+1], y = ru[(size_t)e*4+2], z = ru[(size_t)e*4+3];
  const float s3 = 1.7320508075688772f, s5 = 2.2360679774997896f, s15 = 3.872983346207417f;
  const float* d = dsh + (size_t)e*9;
  float gx = s3*d[1] + s15*y*d[4] + s15*z*d[7] + s15*x*d[8];
  float gy = s3*d[2] + s15*x*d[4] + s15*z*d[5] - s15*y*d[8];
  float gz = s3*d[3] + s15*y*d[5] + 3.f*s5*z*d[6] + s15*x*d[7];
  float dot = gx*x + gy*y + gz*z;
  float iv = 1.f/r;
  float ax = (gx - x*dot)*iv, ay = (gy - y*dot)*iv, az = (gz - z*dot)*iv;
  float xr = r / RMAXC, env = 0.f, denv = 0.f;
  if (xr < 1.f){
    float x2=xr*xr, x3=x2*xr, x5=x2*x3, x6=x5*xr, x7=x6*xr, x8=x7*xr;
    env  = 1.f - 28.f*x6 + 48.f*x7 - 21.f*x8;
    denv = (-168.f*x5 + 336.f*x6 - 168.f*x7)/RMAXC;
  }
  float cc = sqrtf(2.f/RMAXC), S = 0.f;
  #pragma unroll
  for (int b=0;b<8;b++){
    float nn = (float)(b+1), a = nn*PI_F/RMAXC;
    float sn = __sinf(a*r), cs = __cosf(a*r);
    float bess  = cc*sn*iv;
    float dbess = cc*(a*cs*iv - sn*iv*iv);
    S += defv[(size_t)e*16+b]*(dbess*env + bess*denv);
  }
  float dvx = ax + S*x, dvy = ay + S*y, dvz = az + S*z;
  int s = ei[e], rc = ei[E+e];
  atomicAdd(&F[rc*3+0], -dvx); atomicAdd(&F[rc*3+1], -dvy); atomicAdd(&F[rc*3+2], -dvz);
  atomicAdd(&F[s*3+0],   dvx); atomicAdd(&F[s*3+1],   dvy); atomicAdd(&F[s*3+2],   dvz);
}

// ---------------------------------------------------------------------------
extern "C" void kernel_launch(void* const* d_in, const int* in_sizes, int n_in,
                              void* d_out, int out_size, void* d_ws, size_t ws_size,
                              hipStream_t stream)
{
  (void)in_sizes; (void)n_in; (void)out_size; (void)ws_size;
  const int E = NEDGE, NN = NNODE;

  const float* positions = (const float*)d_in[0];
  const float* node_attrs= (const float*)d_in[1];
  const float* shifts    = (const float*)d_in[2];
  const float* atomic_e  = (const float*)d_in[3];
  const float* W_embed   = (const float*)d_in[4];
  const float* RW[2][4]  = {{(const float*)d_in[5],(const float*)d_in[6],(const float*)d_in[7],(const float*)d_in[8]},
                            {(const float*)d_in[9],(const float*)d_in[10],(const float*)d_in[11],(const float*)d_in[12]}};
  const float* Wlin[2]   = {(const float*)d_in[13],(const float*)d_in[17]};
  const float* Wsc[2]    = {(const float*)d_in[14],(const float*)d_in[18]};
  const float* Wprod[2]  = {(const float*)d_in[15],(const float*)d_in[19]};
  const float* wread1    = (const float*)d_in[16];
  const float* Wr1       = (const float*)d_in[20];
  const float* wr2       = (const float*)d_in[21];
  const int*   ei        = (const int*)d_in[22];
  const int*   batch     = (const int*)d_in[23];

  float* outF   = (float*)d_out;
  float* totE   = outF;                 // 32
  float* nodeE  = outF + 32;            // 10000
  float* forces = outF + 10032;         // 30000
  float* nfo    = outF + 40032;         // 10000*1152

  char* base = (char*)d_ws; size_t off = 0;
  auto alloc = [&](size_t bytes)->void*{
    off = (off + 255) & ~(size_t)255; void* p = base + off; off += bytes; return p;
  };
  float* sh    = (float*)alloc((size_t)E*9*4);
  float* ru    = (float*)alloc((size_t)E*4*4);
  float* agg   = (float*)alloc((size_t)NN*576*4);
  float* feats1= (float*)alloc((size_t)NN*576*4);
  float* feats2= (float*)alloc((size_t)NN*576*4);
  float* p1    = (float*)alloc((size_t)NN*64*4);
  float* p2    = (float*)alloc((size_t)NN*64*4);
  float* u2    = (float*)alloc((size_t)NN*16*4);
  float* e0n   = (float*)alloc((size_t)NN*4);
  float* e1n   = (float*)alloc((size_t)NN*4);
  float* e2n   = (float*)alloc((size_t)NN*4);
  float* gnode = (float*)alloc((size_t)NN*64*4);
  float* dinv  = (float*)alloc((size_t)NN*64*4);
  float* ds1   = (float*)alloc((size_t)NN*64*4);
  float* dagg  = (float*)alloc((size_t)NN*576*4);
  float* dshb  = (float*)alloc((size_t)E*9*4);
  float* defv  = (float*)alloc((size_t)E*16*4);

  h16* efp  = (h16*)alloc((size_t)E*32*2);
  h16* h0f  = (h16*)alloc((size_t)NN*64*2);
  h16* s1   = (h16*)alloc((size_t)NN*64*2);
  h16* s2   = (h16*)alloc((size_t)NN*64*2);
  h16* z1[2], *z2[2], *z3[2];
  for (int L2=0; L2<2; L2++){
    z1[L2] = (h16*)alloc((size_t)E*64*2);
    z2[L2] = (h16*)alloc((size_t)E*64*2);
    z3[L2] = (h16*)alloc((size_t)E*64*2);
  }
  h16* dzA  = (h16*)alloc((size_t)E*64*2);
  h16* dzB  = (h16*)alloc((size_t)E*64*2);
  h16* aggT = (h16*)alloc((size_t)9*NN*64*2);
  h16* dfT  = (h16*)alloc((size_t)9*NN*64*2);
  h16* invh = (h16*)alloc((size_t)NN*64*2);
  h16* dph  = (h16*)alloc((size_t)NN*64*2);
  h16 *Bt1[2],*Bt2[2],*Bt3[2],*Bt4[2],*W4r[2],*W3r[2],*W2r[2],*W1r[2];
  h16 *Btlin[2],*Wlinr[2],*Btsc[2],*Wscr[2],*Btpr[2],*Wprr[2];
  for (int L2=0; L2<2; L2++){
    Bt1[L2]=(h16*)alloc(64*32*2);  Bt2[L2]=(h16*)alloc(4096*2);
    Bt3[L2]=(h16*)alloc(4096*2);   Bt4[L2]=(h16*)alloc(576*64*2);
    W4r[L2]=(h16*)alloc(576*64*2); W3r[L2]=(h16*)alloc(4096*2);
    W2r[L2]=(h16*)alloc(4096*2);   W1r[L2]=(h16*)alloc(16*64*2);
    Btlin[L2]=(h16*)alloc(9*4096*2); Wlinr[L2]=(h16*)alloc(9*4096*2);
    Btsc[L2]=(h16*)alloc(4096*2);  Wscr[L2]=(h16*)alloc(4096*2);
    Btpr[L2]=(h16*)alloc(4096*2);  Wprr[L2]=(h16*)alloc(4096*2);
  }
  h16* Btr1 = (h16*)alloc(16*64*2);

  auto zero = [&](float* p, long long n){
    k_zero<<<(unsigned)((n+255)/256),256,0,stream>>>(p,n);
  };
  auto cvt  = [&](const float* s, h16* d, int n){
    k_cvt<<<(n+255)/256,256,0,stream>>>(s,d,n);
  };
  auto cvtT = [&](const float* s, h16* d, int R, int C, int dR){
    int n=C*dR; k_cvt_t<<<(n+255)/256,256,0,stream>>>(s,d,R,C,dR);
  };
  auto cvtP = [&](const float* s, h16* d, int R, int C, int dR){
    int n=dR*C; k_cvt_padR<<<(n+255)/256,256,0,stream>>>(s,d,R,C,dR);
  };
  auto gemm = [&](const h16* A, const h16* Bt, void* Out, const h16* Zg,
                  int M,int K,int N,int lda,int ldb,int ldo,int oes,int ooff,int flags){
    int tiles = (M>>4)*(N>>4);
    k_gemm<<<(tiles+3)/4,128,0,stream>>>(A,Bt,Out,Zg,M,K,N,lda,ldb,ldo,oes,ooff,flags);
  };

  zero(outF, 32LL + NN + NN*3 + (long long)NN*1152);
  zero(dshb, (long long)E*9);

  for (int L2=0; L2<2; L2++){
    cvtT(RW[L2][0], Bt1[L2], 8, 64, 32);
    cvtT(RW[L2][1], Bt2[L2], 64, 64, 64);
    cvtT(RW[L2][2], Bt3[L2], 64, 64, 64);
    cvtT(RW[L2][3], Bt4[L2], 64, 576, 64);
    cvt (RW[L2][3], W4r[L2], 64*576);
    cvt (RW[L2][2], W3r[L2], 4096);
    cvt (RW[L2][1], W2r[L2], 4096);
    cvtP(RW[L2][0], W1r[L2], 8, 64, 16);
    for (int l=0;l<9;l++) cvtT(Wlin[L2]+l*4096, Btlin[L2]+l*4096, 64,64,64);
    cvt (Wlin[L2], Wlinr[L2], 9*4096);
    cvtT(Wsc[L2],  Btsc[L2], 64,64,64);  cvt(Wsc[L2],  Wscr[L2], 4096);
    cvtT(Wprod[L2],Btpr[L2], 64,64,64);  cvt(Wprod[L2],Wprr[L2], 4096);
  }
  cvtT(Wr1, Btr1, 64, 16, 64);

  k_geom<<<(E+255)/256,256,0,stream>>>(positions, shifts, ei, sh, ru, efp, E);
  k_embed<<<(NN*64+255)/256,256,0,stream>>>(node_attrs, W_embed, atomic_e, batch, h0f, e0n, totE, NN);

  // ======================= interaction layers (forward) ====================
  for (int L2=0; L2<2; L2++){
    const h16* nodef  = (L2==0) ? h0f : s1;
    float*     feats  = (L2==0) ? feats1 : feats2;
    float*     pbuf   = (L2==0) ? p1 : p2;
    h16*       sOut   = (L2==0) ? s1 : s2;

    gemm(efp,    Bt1[L2], z1[L2], nullptr, E, 32, 64, 32, 32, 64,1,0, GF_OUTF16);
    gemm(z1[L2], Bt2[L2], z2[L2], nullptr, E, 64, 64, 64, 64, 64,1,0, GF_OUTF16|GF_SILUA);
    gemm(z2[L2], Bt3[L2], z3[L2], nullptr, E, 64, 64, 64, 64, 64,1,0, GF_OUTF16|GF_SILUA);

    zero(agg, (long long)NN*576);
    k_msgagg<<<2048,128,0,stream>>>(z3[L2], Bt4[L2], nodef, sh, ei, agg, E);

    k_aggT<<<(NN*576+255)/256,256,0,stream>>>(agg, aggT, NN);
    for (int l=0;l<9;l++)
      gemm(aggT+(size_t)l*NN*64, Btlin[L2]+l*4096, feats, nullptr,
           NN, 64, 64, 64, 64, 576, 9, l, 0);
    gemm(nodef, Btsc[L2], feats, nullptr, NN, 64, 64, 64, 64, 576, 9, 0, GF_ADD);

    k_inv<<<(NN*64+255)/256,256,0,stream>>>(feats, invh, NN);
    gemm(invh, Btpr[L2], pbuf, nullptr, NN, 64, 64, 64, 64, 64, 1, 0, 0);
    k_hprod<<<(NN*64+255)/256,256,0,stream>>>(feats, pbuf, nfo, sOut, NN, L2*576);

    if (L2==0){
      k_read1<<<(NN+255)/256,256,0,stream>>>(s1, wread1, batch, e1n, totE, NN);
    } else {
      gemm(s2, Btr1, u2, nullptr, NN, 64, 16, 64, 64, 16, 1, 0, 0);
      k_read2<<<(NN+255)/256,256,0,stream>>>(u2, wr2, e0n, e1n, batch, e2n, nodeE, totE, NN);
    }
  }

  // ============================ backward (forces) ==========================
  k_bwd_read<<<(NN*64+255)/256,256,0,stream>>>(u2, wr2, Wr1, feats2, gnode, dph, NN);
  gemm(dph, Wprr[1], dinv, nullptr, NN, 64, 64, 64, 64, 64, 1, 0, 0);
  k_dfeats<<<(NN*64+255)/256,256,0,stream>>>(feats2, p2, gnode, dinv, dfT, NN);
  for (int l=0;l<9;l++)
    gemm(dfT+(size_t)l*NN*64, Wlinr[1]+l*4096, dagg, nullptr,
         NN, 64, 64, 64, 64, 576, 9, l, 0);
  gemm(dfT, Wscr[1], ds1, nullptr, NN, 64, 64, 64, 64, 64, 1, 0, 0);

  k_edgebwd<<<2048,128,0,stream>>>(z3[1], Bt4[1], W4r[1], s1, sh, dagg, ei, dshb, ds1, dzA, E);
  gemm(dzA, W3r[1], dzB, z2[1], E, 64, 64, 64, 64, 64, 1, 0, GF_OUTF16|GF_GRADZ);
  gemm(dzB, W2r[1], dzA, z1[1], E, 64, 64, 64, 64, 64, 1, 0, GF_OUTF16|GF_GRADZ);
  gemm(dzA, W1r[1], defv, nullptr, E, 64, 16, 64, 64, 16, 1, 0, 0);

  k_g1<<<(NN*64+255)/256,256,0,stream>>>(ds1, wread1, feats1, gnode, dph, NN);
  gemm(dph, Wprr[0], dinv, nullptr, NN, 64, 64, 64, 64, 64, 1, 0, 0);
  k_dfeats<<<(NN*64+255)/256,256,0,stream>>>(feats1, p1, gnode, dinv, dfT, NN);
  for (int l=0;l<9;l++)
    gemm(dfT+(size_t)l*NN*64, Wlinr[0]+l*4096, dagg, nullptr,
         NN, 64, 64, 64, 64, 576, 9, l, 0);

  k_edgebwd<<<2048,128,0,stream>>>(z3[0], Bt4[0], W4r[0], h0f, sh, dagg, ei, dshb, nullptr, dzA, E);
  gemm(dzA, W3r[0], dzB, z2[0], E, 64, 64, 64, 64, 64, 1, 0, GF_OUTF16|GF_GRADZ);
  gemm(dzB, W2r[0], dzA, z1[0], E, 64, 64, 64, 64, 64, 1, 0, GF_OUTF16|GF_GRADZ);
  gemm(dzA, W1r[0], defv, nullptr, E, 64, 16, 64, 64, 16, 1, 0, GF_ADD);

  k_force<<<(E+255)/256,256,0,stream>>>(ru, dshb, defv, ei, forces, E);
}
